// manyToManyRNN_88622355186396
// MI455X (gfx1250) — matrix operand order repaired and checked by reading, not compiled, not run on hardware
//
#include <hip/hip_runtime.h>
#include <stdint.h>

#define B_ 64
#define T_ 256
#define I_ 1024
#define H_ 2048
#define O_ 1024
#define LDSTR 2056   // LDS row stride in elements: 2048 + 4-bank shift padding

typedef __bf16 v16bf __attribute__((ext_vector_type(16)));
typedef float  v8f   __attribute__((ext_vector_type(8)));
typedef unsigned short u16b;   // raw bf16 bits

union FragU { v16bf v; uint4 u[2]; };

__device__ __forceinline__ v8f zero8() {
    v8f z = {0.f, 0.f, 0.f, 0.f, 0.f, 0.f, 0.f, 0.f};
    return z;
}

// f32 -> bf16 bits, round-to-nearest-even
__device__ __forceinline__ u16b f2bf(float x) {
    uint32_t u = __float_as_uint(x);
    uint32_t r = u + 0x7fffu + ((u >> 16) & 1u);
    return (u16b)(r >> 16);
}

__device__ __forceinline__ uint32_t pack2bf(float lo, float hi) {
    return (uint32_t)f2bf(lo) | ((uint32_t)f2bf(hi) << 16);
}

__device__ __forceinline__ float sigm(float x) {
    return 1.0f / (1.0f + __expf(-x));
}

// A-matrix fragment (16x32 bf16): lane holds row m=lane%16,
// K = {half*8 .. +8} U {16+half*8 .. +8}
__device__ __forceinline__ v16bf load_frag_a(const u16b* rowptr, int k0, int half) {
    FragU f;
    f.u[0] = *(const uint4*)(rowptr + k0 + half * 8);
    f.u[1] = *(const uint4*)(rowptr + k0 + 16 + half * 8);
    return f.v;
}

// B-matrix fragment (32x16 bf16), B(k,n) = W[n,k]: lane holds W-row n=lane%16,
// K = {half*8 .. +8} U {16+half*8 .. +8}
__device__ __forceinline__ v16bf load_frag_b(const u16b* rowptr, int k0, int half) {
    FragU f;
    const u16b* p = rowptr + k0 + half * 8;
    f.u[0] = *(const uint4*)(p);
    f.u[1] = *(const uint4*)(p + 16);
    return f.v;
}

__device__ __forceinline__ v8f wmma_bf16(v16bf a, v16bf b, v8f c) {
    return __builtin_amdgcn_wmma_f32_16x16x32_bf16(
        /*neg_a=*/false, a, /*neg_b=*/false, b,
        /*c_mod=*/(short)0, c, /*reuse_a=*/false, /*reuse_b=*/false);
}

// ---------------------------------------------------------------- converts
// n must be a multiple of 8 (all our sizes are multiples of 1024)
__global__ void cvt_f32_to_bf16(const float* __restrict__ src,
                                u16b* __restrict__ dst, int n) {
    const float4* s4 = (const float4*)src;
    uint4* d4 = (uint4*)dst;
    const int n8 = n >> 3;
    for (int i = blockIdx.x * blockDim.x + threadIdx.x; i < n8;
         i += gridDim.x * blockDim.x) {
        float4 x0 = s4[2 * i];
        float4 x1 = s4[2 * i + 1];
        uint4 o;
        o.x = pack2bf(x0.x, x0.y);
        o.y = pack2bf(x0.z, x0.w);
        o.z = pack2bf(x1.x, x1.y);
        o.w = pack2bf(x1.z, x1.w);
        d4[i] = o;
    }
}

__global__ void init_barrier(uint32_t* bar) { *bar = 0u; }

// ---------------------------------------------------------------- phase 1
// Z[t*B+b, h] = x[b,t,:] . W_ax[h,:] + b_aa[h] + b_ax[h]
// M = T*B = 16384 (row m -> x row b*T + t), N = H = 2048, K = I = 1024
// one wave computes a 32x64 strip (2 M-tiles x 4 N-tiles), software-pipelined
__global__ void __launch_bounds__(256)
gemm_xw_z(const u16b* __restrict__ Xb, const u16b* __restrict__ Wax,
          const float* __restrict__ baa, const float* __restrict__ bax,
          float* __restrict__ Z) {
    const int lane = threadIdx.x & 31;
    const int half = lane >> 4;
    const int col  = lane & 15;
    const int wid  = blockIdx.x * 8 + (threadIdx.x >> 5);
    const int mgrp = wid >> 5;    // 0..511
    const int ngrp = wid & 31;    // 0..31
    const int M0 = mgrp << 5;     // 32 rows
    const int N0 = ngrp << 6;     // 64 cols

    const u16b* arowp[2];
#pragma unroll
    for (int p = 0; p < 2; ++p) {
        const int m_a = M0 + p * 16 + col;
        const int tt = m_a >> 6;   // m / B
        const int bb = m_a & 63;   // m % B
        arowp[p] = Xb + ((size_t)bb * T_ + tt) * I_;
    }
    const u16b* browp[4];
#pragma unroll
    for (int q = 0; q < 4; ++q)
        browp[q] = Wax + (size_t)(N0 + q * 16 + col) * I_;

    v8f acc[2][4];
#pragma unroll
    for (int p = 0; p < 2; ++p)
#pragma unroll
        for (int q = 0; q < 4; ++q) acc[p][q] = zero8();

    v16bf af0[2], af1[2], bf0[4], bf1[4];
#pragma unroll
    for (int p = 0; p < 2; ++p) af0[p] = load_frag_a(arowp[p], 0, half);
#pragma unroll
    for (int q = 0; q < 4; ++q) bf0[q] = load_frag_b(browp[q], 0, half);

    int k0 = 0;
    for (; k0 < I_ - 64; k0 += 64) {
#pragma unroll
        for (int p = 0; p < 2; ++p) af1[p] = load_frag_a(arowp[p], k0 + 32, half);
#pragma unroll
        for (int q = 0; q < 4; ++q) bf1[q] = load_frag_b(browp[q], k0 + 32, half);
#pragma unroll
        for (int p = 0; p < 2; ++p)
#pragma unroll
            for (int q = 0; q < 4; ++q)
                acc[p][q] = wmma_bf16(af0[p], bf0[q], acc[p][q]);
#pragma unroll
        for (int p = 0; p < 2; ++p) af0[p] = load_frag_a(arowp[p], k0 + 64, half);
#pragma unroll
        for (int q = 0; q < 4; ++q) bf0[q] = load_frag_b(browp[q], k0 + 64, half);
#pragma unroll
        for (int p = 0; p < 2; ++p)
#pragma unroll
            for (int q = 0; q < 4; ++q)
                acc[p][q] = wmma_bf16(af1[p], bf1[q], acc[p][q]);
    }
    // k0 == I_-64: last two K-steps
#pragma unroll
    for (int p = 0; p < 2; ++p) af1[p] = load_frag_a(arowp[p], k0 + 32, half);
#pragma unroll
    for (int q = 0; q < 4; ++q) bf1[q] = load_frag_b(browp[q], k0 + 32, half);
#pragma unroll
    for (int p = 0; p < 2; ++p)
#pragma unroll
        for (int q = 0; q < 4; ++q)
            acc[p][q] = wmma_bf16(af0[p], bf0[q], acc[p][q]);
#pragma unroll
    for (int p = 0; p < 2; ++p)
#pragma unroll
        for (int q = 0; q < 4; ++q)
            acc[p][q] = wmma_bf16(af1[p], bf1[q], acc[p][q]);

#pragma unroll
    for (int q = 0; q < 4; ++q) {
        const int n = N0 + q * 16 + col;
        const float bsum = baa[n] + bax[n];
#pragma unroll
        for (int p = 0; p < 2; ++p)
#pragma unroll
            for (int j = 0; j < 8; ++j) {
                const int m = M0 + p * 16 + j + 8 * half;
                Z[(size_t)m * H_ + n] = acc[p][q][j] + bsum;
            }
    }
}

// ---------------------------------------------------------------- phase 2
// Persistent recurrence: a_t = sigmoid(a_{t-1} . W_aa^T + Z[t])
// 64 blocks x 256 threads. Block owns 32 columns of W_aa, resident in LDS.
// 8 waves = 4 M-tiles x 2 N-tiles, each wave one 16x16 tile, full K=2048,
// software-pipelined (global A frags + LDS B frags). Global atomic barrier.
__global__ void __launch_bounds__(256)
rnn_rec(const u16b* __restrict__ Waa, const float* __restrict__ Z,
        u16b* __restrict__ Aall, uint32_t* __restrict__ bar) {
    extern __shared__ char smem_raw[];
    u16b* wlds = (u16b*)smem_raw;

    const int N0blk = blockIdx.x * 32;

    // stage 32 rows x 2048 cols of W_aa (bf16) into LDS, padded rows
    for (int c = threadIdx.x; c < 32 * (H_ / 8); c += blockDim.x) {
        int row = c >> 8;     // H_/8 = 256 chunks per row
        int cc  = c & 255;
        *(uint4*)(wlds + row * LDSTR + cc * 8) =
            *(const uint4*)(Waa + (size_t)(N0blk + row) * H_ + cc * 8);
    }
    __syncthreads();

    const int lane = threadIdx.x & 31;
    const int half = lane >> 4;
    const int col  = lane & 15;
    const int wv   = threadIdx.x >> 5;
    const int m_t  = wv & 3;      // M-tile 0..3  (B = 64 rows)
    const int n_t  = wv >> 2;     // N-tile 0..1
    const u16b* wrow = wlds + (size_t)(n_t * 16 + col) * LDSTR;
    const int mrow = m_t * 16 + col;
    const int n = N0blk + n_t * 16 + col;
    const uint32_t nblocks = gridDim.x;

    for (int t = 0; t < T_; ++t) {
        // hoist Z loads: overlap with the K reduction below
        const float* zcol = Z + (size_t)t * B_ * H_ + n;
        float zv[8];
#pragma unroll
        for (int j = 0; j < 8; ++j)
            zv[j] = zcol[(size_t)(m_t * 16 + j + 8 * half) * H_];

        v8f acc = zero8();
        if (t > 0) {
            const u16b* arow = Aall + ((size_t)(t - 1) * B_ + mrow) * H_;
            v16bf af0 = load_frag_a(arow, 0, half);
            v16bf bw0 = load_frag_b(wrow, 0, half);
            int k0 = 0;
            for (; k0 < H_ - 64; k0 += 64) {
                v16bf af1 = load_frag_a(arow, k0 + 32, half);
                v16bf bw1 = load_frag_b(wrow, k0 + 32, half);
                acc = wmma_bf16(af0, bw0, acc);
                af0 = load_frag_a(arow, k0 + 64, half);
                bw0 = load_frag_b(wrow, k0 + 64, half);
                acc = wmma_bf16(af1, bw1, acc);
            }
            v16bf af1 = load_frag_a(arow, k0 + 32, half);
            v16bf bw1 = load_frag_b(wrow, k0 + 32, half);
            acc = wmma_bf16(af0, bw0, acc);
            acc = wmma_bf16(af1, bw1, acc);
        }

        u16b* abase = Aall + (size_t)t * B_ * H_;
#pragma unroll
        for (int j = 0; j < 8; ++j) {
            const int m = m_t * 16 + j + 8 * half;
            abase[(size_t)m * H_ + n] = f2bf(sigm(acc[j] + zv[j]));
        }

        if (t + 1 < T_) {
            __syncthreads();
            if (threadIdx.x == 0) {
                __threadfence();
                atomicAdd(bar, 1u);
                const uint32_t target = nblocks * (uint32_t)(t + 1);
                while (__hip_atomic_load(bar, __ATOMIC_ACQUIRE,
                                         __HIP_MEMORY_SCOPE_AGENT) < target) {
                    __builtin_amdgcn_s_sleep(1);
                }
            }
            __syncthreads();
        }
    }
}

// ---------------------------------------------------------------- phase 3
// Y[m, o] = sigmoid(A[m,:] . W_ya[o,:] + b_ya[o]),  m = t*B+b  (matches
// reference's raw reshape of the time-major scan output)
// one wave computes a 32x64 strip (2 M-tiles x 4 N-tiles), software-pipelined
__global__ void __launch_bounds__(256)
gemm_ay(const u16b* __restrict__ Ab, const u16b* __restrict__ Wya,
        const float* __restrict__ bya, float* __restrict__ Y) {
    const int lane = threadIdx.x & 31;
    const int half = lane >> 4;
    const int col  = lane & 15;
    const int wid  = blockIdx.x * 8 + (threadIdx.x >> 5);
    const int mgrp = wid >> 4;    // 0..511
    const int ngrp = wid & 15;    // 0..15
    const int M0 = mgrp << 5;     // 32 rows
    const int N0 = ngrp << 6;     // 64 cols

    const u16b* arowp[2];
#pragma unroll
    for (int p = 0; p < 2; ++p)
        arowp[p] = Ab + (size_t)(M0 + p * 16 + col) * H_;
    const u16b* browp[4];
#pragma unroll
    for (int q = 0; q < 4; ++q)
        browp[q] = Wya + (size_t)(N0 + q * 16 + col) * H_;

    v8f acc[2][4];
#pragma unroll
    for (int p = 0; p < 2; ++p)
#pragma unroll
        for (int q = 0; q < 4; ++q) acc[p][q] = zero8();

    v16bf af0[2], af1[2], bf0[4], bf1[4];
#pragma unroll
    for (int p = 0; p < 2; ++p) af0[p] = load_frag_a(arowp[p], 0, half);
#pragma unroll
    for (int q = 0; q < 4; ++q) bf0[q] = load_frag_b(browp[q], 0, half);

    int k0 = 0;
    for (; k0 < H_ - 64; k0 += 64) {
#pragma unroll
        for (int p = 0; p < 2; ++p) af1[p] = load_frag_a(arowp[p], k0 + 32, half);
#pragma unroll
        for (int q = 0; q < 4; ++q) bf1[q] = load_frag_b(browp[q], k0 + 32, half);
#pragma unroll
        for (int p = 0; p < 2; ++p)
#pragma unroll
            for (int q = 0; q < 4; ++q)
                acc[p][q] = wmma_bf16(af0[p], bf0[q], acc[p][q]);
#pragma unroll
        for (int p = 0; p < 2; ++p) af0[p] = load_frag_a(arowp[p], k0 + 64, half);
#pragma unroll
        for (int q = 0; q < 4; ++q) bf0[q] = load_frag_b(browp[q], k0 + 64, half);
#pragma unroll
        for (int p = 0; p < 2; ++p)
#pragma unroll
            for (int q = 0; q < 4; ++q)
                acc[p][q] = wmma_bf16(af1[p], bf1[q], acc[p][q]);
    }
#pragma unroll
    for (int p = 0; p < 2; ++p) af1[p] = load_frag_a(arowp[p], k0 + 32, half);
#pragma unroll
    for (int q = 0; q < 4; ++q) bf1[q] = load_frag_b(browp[q], k0 + 32, half);
#pragma unroll
    for (int p = 0; p < 2; ++p)
#pragma unroll
        for (int q = 0; q < 4; ++q)
            acc[p][q] = wmma_bf16(af0[p], bf0[q], acc[p][q]);
#pragma unroll
    for (int p = 0; p < 2; ++p)
#pragma unroll
        for (int q = 0; q < 4; ++q)
            acc[p][q] = wmma_bf16(af1[p], bf1[q], acc[p][q]);

#pragma unroll
    for (int q = 0; q < 4; ++q) {
        const int n = N0 + q * 16 + col;
        const float bv = bya[n];
#pragma unroll
        for (int p = 0; p < 2; ++p)
#pragma unroll
            for (int j = 0; j < 8; ++j) {
                const int m = M0 + p * 16 + j + 8 * half;
                Y[(size_t)m * O_ + n] = sigm(acc[p][q][j] + bv);
            }
    }
}

// ----------------------------------------------------------------
extern "C" void kernel_launch(void* const* d_in, const int* in_sizes, int n_in,
                              void* d_out, int out_size, void* d_ws, size_t ws_size,
                              hipStream_t stream) {
    (void)in_sizes; (void)n_in; (void)out_size; (void)ws_size;
    const float* x   = (const float*)d_in[0];
    const float* Waa = (const float*)d_in[1];
    const float* baa = (const float*)d_in[2];
    const float* Wax = (const float*)d_in[3];
    const float* bax = (const float*)d_in[4];
    const float* Wya = (const float*)d_in[5];
    const float* bya = (const float*)d_in[6];

    char* ws = (char*)d_ws;
    size_t off = 0;
    auto take = [&](size_t bytes) -> char* {
        char* p = ws + off;
        off = (off + bytes + 255) & ~(size_t)255;
        return p;
    };
    u16b* Waa_b = (u16b*)take((size_t)H_ * H_ * 2);       // 8 MB
    u16b* Wax_b = (u16b*)take((size_t)H_ * I_ * 2);       // 4 MB
    u16b* Wya_b = (u16b*)take((size_t)O_ * H_ * 2);       // 4 MB
    u16b* X_b   = (u16b*)take((size_t)B_ * T_ * I_ * 2);  // 32 MB
    float* Z    = (float*)take((size_t)T_ * B_ * H_ * 4); // 128 MB
    u16b* A_b   = (u16b*)take((size_t)T_ * B_ * H_ * 2);  // 64 MB
    uint32_t* bar = (uint32_t*)take(256);

    cvt_f32_to_bf16<<<1024, 256, 0, stream>>>(Waa, Waa_b, H_ * H_);
    cvt_f32_to_bf16<<<1024, 256, 0, stream>>>(Wax, Wax_b, H_ * I_);
    cvt_f32_to_bf16<<<1024, 256, 0, stream>>>(Wya, Wya_b, O_ * H_);
    cvt_f32_to_bf16<<<2048, 256, 0, stream>>>(x, X_b, B_ * T_ * I_);
    init_barrier<<<1, 1, 0, stream>>>(bar);

    // Z = x . W_ax^T + b  (16384x2048x1024): 512 Mgroups x 32 Ngroups, 8 waves/blk
    gemm_xw_z<<<2048, 256, 0, stream>>>(X_b, Wax_b, baa, bax, Z);

    // persistent recurrence: 64 blocks, W_aa slab resident in LDS (~128.5 KB)
    rnn_rec<<<64, 256, (size_t)(32 * LDSTR * 2), stream>>>(Waa_b, Z, A_b, bar);

    // y = sigmoid(A . W_ya^T + b_ya) (16384x1024x2048)
    gemm_ay<<<1024, 256, 0, stream>>>(A_b, Wya_b, bya, (float*)d_out);
}
